// MLP_22771916603452
// MI455X (gfx1250) — matrix operand order repaired and checked by reading, not compiled, not run on hardware
//
#include <hip/hip_runtime.h>

typedef __attribute__((ext_vector_type(16))) _Float16 v16h;
typedef __attribute__((ext_vector_type(2)))  _Float16 h2;
typedef __attribute__((ext_vector_type(8)))  float    v8f;
typedef __attribute__((ext_vector_type(4)))  float    v4f;
typedef __attribute__((ext_vector_type(8)))  int      v8i;

#define N_INPUT   32
#define N_NEURONS 64
#define N_OUTPUT  16
#define BATCH     2000000
#define TILES     (BATCH / 16)        // 125000 batch tiles of 16 rows

#define WAVES_PER_BLOCK 8
#define BLOCK_THREADS   (WAVES_PER_BLOCK * 32)
#define GRID_BLOCKS     640

#define SWZ_SWAPX16 0x401F            // ds_swizzle: xor=0x10, or=0, and=0x1f

#define WMMA(A, B, C) __builtin_amdgcn_wmma_f32_16x16x32_f16( \
    false, (A), false, (B), (short)0, (C), false, false)

// Single-instruction ReLU: v_med3_f32(x, 0, +inf) -- avoids the IEEE-mode
// canonicalize (max(x,x)) that clang inserts when lowering maxnum(x, 0).
__device__ __forceinline__ float relu(float x) {
  return __builtin_amdgcn_fmed3f(x, 0.0f, __builtin_inff());
}

// Pack two f32 (with ReLU) into one dword of f16 pairs (v_cvt_pk_f16_f32).
__device__ __forceinline__ int pack_relu(float a, float b) {
  h2 p; p[0] = (_Float16)relu(a); p[1] = (_Float16)relu(b);
  return __builtin_bit_cast(int, p);
}

// 16 f32 (4x v4f) -> v16h in ascending element order.
__device__ __forceinline__ v16h cvt16(const v4f a, const v4f b,
                                      const v4f c, const v4f d) {
  v16h r;
  r[0]  = (_Float16)a[0]; r[1]  = (_Float16)a[1]; r[2]  = (_Float16)a[2]; r[3]  = (_Float16)a[3];
  r[4]  = (_Float16)b[0]; r[5]  = (_Float16)b[1]; r[6]  = (_Float16)b[2]; r[7]  = (_Float16)b[3];
  r[8]  = (_Float16)c[0]; r[9]  = (_Float16)c[1]; r[10] = (_Float16)c[2]; r[11] = (_Float16)c[3];
  r[12] = (_Float16)d[0]; r[13] = (_Float16)d[1]; r[14] = (_Float16)d[2]; r[15] = (_Float16)d[3];
  return r;
}

// A operand (16x32 f16, M x K) built from one weight row, K-chunk base p:
// lane<16: M=lane, K = 0..15; lane>=16: M=lane-16, K = 16..31.
__device__ __forceinline__ v16h a_op(const float* __restrict__ p, int h) {
  const v4f* q = (const v4f*)(p + 16 * h);
  return cvt16(q[0], q[1], q[2], q[3]);
}

// Build one activation B operand (32x16 f16, K x N) from two D tiles whose
// M ranges are [32c..32c+15] (tlo) and [32c+16..32c+31] (thi), with ReLU.
// D layout: VGPR r -> row M=r+8h, lane -> col N.  B layout used here (matches a_op): VGPR v -> K=2v,2v+1
// for lanes<16 (K+16 for lanes>=16).  The M=8..15 rows sit in the opposite
// 16-lane half -> one SWAPX16 ds_swizzle + cndmask per register.
__device__ __forceinline__ v16h b_from_d(const v8f tlo, const v8f thi, bool lo) {
  v8i b;
  #pragma unroll
  for (int v = 0; v < 4; ++v) {
    const int plo = pack_relu(tlo[2 * v], tlo[2 * v + 1]);   // M = 2v,2v+1 (+8 in hi half)
    const int phi = pack_relu(thi[2 * v], thi[2 * v + 1]);
    const int slo = __builtin_amdgcn_ds_swizzle(plo, SWZ_SWAPX16);
    const int shi = __builtin_amdgcn_ds_swizzle(phi, SWZ_SWAPX16);
    b[v]     = lo ? plo : shi;   // K rel 2v,2v+1   (lanes>=16: abs K=16+2v -> thi rows 0..7)
    b[v + 4] = lo ? slo : phi;   // K rel 8+2v      (lanes>=16: abs K=24+2v -> thi rows 8..15)
  }
  return __builtin_bit_cast(v16h, b);
}

__global__ __launch_bounds__(BLOCK_THREADS) void mlp_wmma_t_kernel(
    const float* __restrict__ X,    // [BATCH][32]
    const float* __restrict__ W0,   // [64][32]
    const float* __restrict__ W1,   // [64][64]
    const float* __restrict__ W2,   // [16][64]
    float* __restrict__ Out)        // [BATCH][16]
{
  const int lane = threadIdx.x & 31;
  const int wave = threadIdx.x >> 5;
  const int m    = lane & 15;
  const int h    = lane >> 4;
  const bool lo  = lane < 16;

  // ---- Preload all weights once as WMMA A operands (f32 -> f16) ----
  v16h W0A[4];                       // layer0: M=64 -> 4 tiles, K=32
  #pragma unroll
  for (int t = 0; t < 4; ++t)
    W0A[t] = a_op(W0 + (16 * t + m) * N_INPUT, h);

  v16h W1A[4][2];                    // layer1: M=64, K=64 -> 2 K-chunks
  #pragma unroll
  for (int t = 0; t < 4; ++t) {
    #pragma unroll
    for (int c = 0; c < 2; ++c)
      W1A[t][c] = a_op(W1 + (16 * t + m) * N_NEURONS + 32 * c, h);
  }

  v16h W2A[2];                       // layer2: M=16, K=64 -> 2 K-chunks
  #pragma unroll
  for (int c = 0; c < 2; ++c)
    W2A[c] = a_op(W2 + m * N_NEURONS + 32 * c, h);

  const int waveId = blockIdx.x * WAVES_PER_BLOCK + wave;
  const int nWaves = GRID_BLOCKS * WAVES_PER_BLOCK;

  for (int tile = waveId; tile < TILES; tile += nWaves) {
    // ---- x^T as B operand: lane holds batch col m, K = 16h..16h+15 (contig 64B) ----
    const v4f* xr = (const v4f*)(X + (size_t)(tile * 16 + m) * N_INPUT + 16 * h);
    const v16h Bx = cvt16(__builtin_nontemporal_load(xr + 0),
                          __builtin_nontemporal_load(xr + 1),
                          __builtin_nontemporal_load(xr + 2),
                          __builtin_nontemporal_load(xr + 3));

    // prefetch next tile's slice of X
    const int nxt = tile + nWaves;
    if (nxt < TILES)
      __builtin_prefetch(X + (size_t)(nxt * 16 + m) * N_INPUT + 16 * h, 0, 1);

    // ---- Layer 0: y0^T = W0 @ x^T : 4 M-tiles, K=32 ----
    v8f D0[4];
    #pragma unroll
    for (int t = 0; t < 4; ++t) {
      v8f z = {};
      D0[t] = WMMA(W0A[t], Bx, z);
    }

    // ---- ReLU + repack D0 -> B operands (register-only transpose) ----
    v16h B1a = b_from_d(D0[0], D0[1], lo);   // K = 0..31
    v16h B1b = b_from_d(D0[2], D0[3], lo);   // K = 32..63

    // ---- Layer 1: y1^T = W1 @ y0^T : 4 M-tiles, K=64 ----
    v8f D1[4];
    #pragma unroll
    for (int t = 0; t < 4; ++t) {
      v8f z = {};
      v8f acc = WMMA(W1A[t][0], B1a, z);
      D1[t] = WMMA(W1A[t][1], B1b, acc);
    }

    // ---- ReLU + repack D1 -> B operands ----
    v16h B2a = b_from_d(D1[0], D1[1], lo);
    v16h B2b = b_from_d(D1[2], D1[3], lo);

    // ---- Layer 2: out^T = W2 @ y1^T : M=16, K=64 (no ReLU) ----
    v8f z = {};
    v8f c2 = WMMA(W2A[0], B2a, z);
    c2 = WMMA(W2A[1], B2b, c2);

    // ---- Store: lane holds out[batch m][neurons 8h..8h+7] -> 2x b128, coalesced ----
    float* orow = Out + (size_t)(tile * 16 + m) * N_OUTPUT + 8 * h;
    v4f s0 = { c2[0], c2[1], c2[2], c2[3] };
    v4f s1 = { c2[4], c2[5], c2[6], c2[7] };
    __builtin_nontemporal_store(s0, (v4f*)orow);
    __builtin_nontemporal_store(s1, (v4f*)orow + 1);
  }
}

extern "C" void kernel_launch(void* const* d_in, const int* in_sizes, int n_in,
                              void* d_out, int out_size, void* d_ws, size_t ws_size,
                              hipStream_t stream) {
  (void)in_sizes; (void)n_in; (void)out_size; (void)d_ws; (void)ws_size;
  const float* X  = (const float*)d_in[0];
  const float* W0 = (const float*)d_in[1];
  const float* W1 = (const float*)d_in[2];
  const float* W2 = (const float*)d_in[3];
  float* Out = (float*)d_out;
  mlp_wmma_t_kernel<<<GRID_BLOCKS, BLOCK_THREADS, 0, stream>>>(X, W0, W1, W2, Out);
}
